// MultiheadAttention_48258252538315
// MI455X (gfx1250) — matrix operand order repaired and checked by reading, not compiled, not run on hardware
//
#include <hip/hip_runtime.h>

// ---------------- problem constants ----------------
#define BATCH 4
#define SLEN  2048
#define EMB   1024
#define NH    16
#define HD    64          // head dim; note NH*HD == EMB == 1024
#define SM_SCALE 0.125f   // 1/sqrt(64)

typedef __attribute__((ext_vector_type(16))) __bf16 v16bf;
typedef __attribute__((ext_vector_type(8)))  __bf16 v8bf;
typedef __attribute__((ext_vector_type(8)))  float  v8f;
typedef __attribute__((ext_vector_type(4)))  float  v4f;

// ---------------- helpers ----------------
__device__ __forceinline__ v8f zero8() {
    v8f z;
#pragma unroll
    for (int i = 0; i < 8; ++i) z[i] = 0.0f;
    return z;
}

__device__ __forceinline__ v16bf combine16(v8bf a, v8bf b) {
    v16bf r;
#pragma unroll
    for (int i = 0; i < 8; ++i) { r[i] = a[i]; r[i + 8] = b[i]; }
    return r;
}

// A-matrix fragment (16x32 bf16). Lane lr (=lane&15) owns row lr.
// Elements 0..7 are K = lh*8+0..7, elements 8..15 are K = lh*8+16..23.
__device__ __forceinline__ v16bf afrag(const __bf16* rowptr, int k0, int lh) {
    v8bf lo = *(const v8bf*)(rowptr + k0 + lh * 8);
    v8bf hi = *(const v8bf*)(rowptr + k0 + lh * 8 + 16);
    return combine16(lo, hi);
}

// Same A fragment but sourced from fp32 memory (convert on the fly).
__device__ __forceinline__ v16bf afrag_f32(const float* rowptr, int k0, int lh) {
    v4f x0 = *(const v4f*)(rowptr + k0 + lh * 8);
    v4f x1 = *(const v4f*)(rowptr + k0 + lh * 8 + 4);
    v4f x2 = *(const v4f*)(rowptr + k0 + lh * 8 + 16);
    v4f x3 = *(const v4f*)(rowptr + k0 + lh * 8 + 20);
    v16bf r;
#pragma unroll
    for (int i = 0; i < 4; ++i) {
        r[i]      = (__bf16)x0[i];
        r[i + 4]  = (__bf16)x1[i];
        r[i + 8]  = (__bf16)x2[i];
        r[i + 12] = (__bf16)x3[i];
    }
    return r;
}

// B-matrix fragment (32x16 bf16). Lane lr owns column lr; elements 0..7 are
// K = lh*8+0..7, elements 8..15 are K = lh*8+16..23. colptr points at the column's
// K-major (contiguous) data.
__device__ __forceinline__ v16bf bfrag(const __bf16* colptr, int k0, int lh) {
    v8bf b0 = *(const v8bf*)(colptr + k0 + lh * 8);
    v8bf b1 = *(const v8bf*)(colptr + k0 + lh * 8 + 16);
    return combine16(b0, b1);
}

__device__ __forceinline__ v8f wmma_bf16(v16bf a, v16bf b, v8f c) {
    // (neg_a, A, neg_b, B, c_mod, C, reuse_a, reuse_b)
    return __builtin_amdgcn_wmma_f32_16x16x32_bf16(false, a, false, b, (short)0, c,
                                                   false, false);
}

// ---- CDNA5 async global->LDS copy (16B per lane), tracked by ASYNCcnt ----
__device__ __forceinline__ void async_cp16(const void* gptr, void* lptr) {
    // LDS aperture: addr[31:0] is the LDS byte offset (ISA 10.2).
    unsigned l = (unsigned)(uintptr_t)lptr;
    unsigned long long g = (unsigned long long)(uintptr_t)gptr;
    asm volatile("global_load_async_to_lds_b128 %0, %1, off"
                 :: "v"(l), "v"(g) : "memory");
}

__device__ __forceinline__ void wait_async0() {
#if __has_builtin(__builtin_amdgcn_s_wait_asynccnt)
    __builtin_amdgcn_s_wait_asynccnt(0);
#else
    asm volatile("s_wait_asynccnt 0x0" ::: "memory");
#endif
}

// ---------------- kernel 1: weight transposes to bf16 ----------------
// WtQ/K/V: [H][D][E]  (B-frags read contiguous E for output column d)
// WOt:     [E][H*D]   (B-frags read contiguous (h,d) for output column e)
__global__ void cvt_weights(const float* __restrict__ WQ, const float* __restrict__ WK,
                            const float* __restrict__ WV, const float* __restrict__ WO,
                            __bf16* __restrict__ wtq, __bf16* __restrict__ wtk,
                            __bf16* __restrict__ wtv, __bf16* __restrict__ wot) {
    int idx = blockIdx.x * blockDim.x + threadIdx.x;
    const int N = NH * EMB * HD;
    if (idx >= N) return;
    // W_{Q,K,V}: [H][E][D] -> [H][D][E]
    int h = idx / (EMB * HD);
    int r = idx % (EMB * HD);
    int e = r / HD;
    int d = r % HD;
    int t = (h * HD + d) * EMB + e;
    wtq[t] = (__bf16)WQ[idx];
    wtk[t] = (__bf16)WK[idx];
    wtv[t] = (__bf16)WV[idx];
    // W_O: [H][D][E] -> [E][H*D]
    int h2 = idx / (HD * EMB);
    int r2 = idx % (HD * EMB);
    int d2 = r2 / EMB;
    int e2 = r2 % EMB;
    wot[e2 * (NH * HD) + h2 * HD + d2] = (__bf16)WO[idx];
}

// ---------------- kernel 2: per-head input projection ----------------
// act [B,S,E] fp32  x  wt [H][D][E] bf16  (+bias[H,D])
// vmode==0: dst = [B*H, S, D]  (Q, K)
// vmode==1: dst = [B*H, D, S]  (V transposed for OV B-frags)
__global__ __launch_bounds__(128) void proj_kernel(const float* __restrict__ act,
                                                   const __bf16* __restrict__ wt,
                                                   const float* __restrict__ bias,
                                                   __bf16* __restrict__ dst, int vmode) {
    int tid = threadIdx.x;
    int w = tid >> 5, lane = tid & 31, lr = lane & 15, lh = lane >> 4;
    int bh = blockIdx.y;
    int b = bh >> 4, h = bh & 15;
    int sbase = blockIdx.x * 64 + w * 16;

    const float*  arow = act + ((size_t)(b * SLEN + sbase + lr)) * EMB;
    const __bf16* wtb  = wt + h * HD * EMB;

    v8f acc[4];
#pragma unroll
    for (int nt = 0; nt < 4; ++nt) acc[nt] = zero8();

    for (int k0 = 0; k0 < EMB; k0 += 32) {
        v16bf a = afrag_f32(arow, k0, lh);
#pragma unroll
        for (int nt = 0; nt < 4; ++nt) {
            v16bf bm = bfrag(wtb + (nt * 16 + lr) * EMB, k0, lh);
            acc[nt] = wmma_bf16(a, bm, acc[nt]);
        }
    }

#pragma unroll
    for (int nt = 0; nt < 4; ++nt) {
        int d = nt * 16 + lr;
        float bv = bias[h * HD + d];
        if (!vmode) {
#pragma unroll
            for (int v = 0; v < 8; ++v) {
                int s = sbase + v + 8 * lh;
                dst[(bh * SLEN + s) * HD + d] = (__bf16)(acc[nt][v] + bv);
            }
        } else {
            v8bf pk;
#pragma unroll
            for (int v = 0; v < 8; ++v) pk[v] = (__bf16)(acc[nt][v] + bv);
            // s = sbase + 8*lh + v -> contiguous over v; 16B aligned
            *(v8bf*)(dst + (bh * HD + d) * SLEN + sbase + 8 * lh) = pk;
        }
    }
}

// ---------------- kernel 3: flash attention ----------------
// Q,K: [B*H, S, D] bf16 ; Vt: [B*H, D, S] bf16 ; OV out: [B, S, H*D] bf16
// K/V tiles double-buffered in LDS via global_load_async_to_lds_b128.
__global__ __launch_bounds__(128) void attn_kernel(const __bf16* __restrict__ Q,
                                                   const __bf16* __restrict__ K,
                                                   const __bf16* __restrict__ Vt,
                                                   __bf16* __restrict__ OV) {
    __shared__ __align__(16) __bf16 Ktile[2][64][64];  // [buf][key][d]   8KB each
    __shared__ __align__(16) __bf16 Vtile[2][64][64];  // [buf][d][key]   8KB each
    __shared__ __align__(16) __bf16 Pbuf[4][16][72];   // per-wave probs (16B rows)

    int tid = threadIdx.x;
    int w = tid >> 5, lane = tid & 31, lr = lane & 15, lh = lane >> 4;
    int bh = blockIdx.y;
    int b = bh >> 4, h = bh & 15;
    int qbase = blockIdx.x * 64 + w * 16;

    const __bf16* Qb = Q + bh * SLEN * HD;
    const __bf16* Kb = K + bh * SLEN * HD;
    const __bf16* Vb = Vt + bh * HD * SLEN;

    // cooperative async issue of one 64-key tile into buffer `buf`
    auto issue_tile = [&](int kt, int buf) {
        int keybase = kt * 64;
        // K tile: 64 keys x 64 d = contiguous 8192B; 64B per thread
        {
            const __bf16* g = Kb + keybase * HD + tid * 32;
            __bf16* l = &Ktile[buf][0][0] + tid * 32;
#pragma unroll
            for (int c = 0; c < 4; ++c) async_cp16(g + c * 8, l + c * 8);
        }
        // V tile: 64 rows (d) x 128B; 2 threads per row, 64B each
        {
            int r = tid >> 1, h2 = tid & 1;
            const __bf16* g = Vb + r * SLEN + keybase + h2 * 32;
            __bf16* l = &Vtile[buf][r][0] + h2 * 32;
#pragma unroll
            for (int c = 0; c < 4; ++c) async_cp16(g + c * 8, l + c * 8);
        }
    };

    // Q fragments resident for the whole key loop (16 rows x 64 d = 2 A-frags)
    v16bf qf0 = afrag(Qb + (qbase + lr) * HD, 0, lh);
    v16bf qf1 = afrag(Qb + (qbase + lr) * HD, 32, lh);

    float m[8], l[8];
    v8f o[4];
#pragma unroll
    for (int v = 0; v < 8; ++v) { m[v] = -1e30f; l[v] = 0.0f; }
#pragma unroll
    for (int nt = 0; nt < 4; ++nt) o[nt] = zero8();

    issue_tile(0, 0);

    for (int kt = 0; kt < SLEN / 64; ++kt) {
        int cur = kt & 1;
        wait_async0();        // this wave's async stores into buf `cur` landed
        __syncthreads();      // every wave's piece of the tile landed
        if (kt + 1 < SLEN / 64) issue_tile(kt + 1, 1 - cur);  // stream next tile

        // scores S = Q * K^T for a 16x64 tile (B-frags from LDS K tile)
        v8f s4[4];
#pragma unroll
        for (int nt = 0; nt < 4; ++nt) s4[nt] = zero8();
#pragma unroll
        for (int kc = 0; kc < 2; ++kc) {
            v16bf a = (kc == 0) ? qf0 : qf1;
#pragma unroll
            for (int nt = 0; nt < 4; ++nt) {
                v16bf bm = bfrag(&Ktile[cur][nt * 16 + lr][0], kc * 32, lh);
                s4[nt] = wmma_bf16(a, bm, s4[nt]);
            }
        }
#pragma unroll
        for (int nt = 0; nt < 4; ++nt)
#pragma unroll
            for (int v = 0; v < 8; ++v) s4[nt][v] *= SM_SCALE;

        // online softmax (row M = v + 8*lh lives in one 16-lane half -> xor 1,2,4,8)
#pragma unroll
        for (int v = 0; v < 8; ++v) {
            float mx = fmaxf(fmaxf(s4[0][v], s4[1][v]), fmaxf(s4[2][v], s4[3][v]));
#pragma unroll
            for (int off = 1; off < 16; off <<= 1) mx = fmaxf(mx, __shfl_xor(mx, off, 32));
            float nm = fmaxf(m[v], mx);
            float al = __expf(m[v] - nm);
            float ps = 0.0f;
#pragma unroll
            for (int nt = 0; nt < 4; ++nt) {
                float p = __expf(s4[nt][v] - nm);
                ps += p;
                Pbuf[w][v + 8 * lh][nt * 16 + lr] = (__bf16)p;
            }
#pragma unroll
            for (int off = 1; off < 16; off <<= 1) ps += __shfl_xor(ps, off, 32);
            l[v] = l[v] * al + ps;
            m[v] = nm;
#pragma unroll
            for (int nt = 0; nt < 4; ++nt) o[nt][v] *= al;
        }

        // O += P * V  (A-frags of P from per-wave LDS; B-frags from LDS V tile)
#pragma unroll
        for (int kc = 0; kc < 2; ++kc) {
            v16bf pa = afrag(&Pbuf[w][lr][0], kc * 32, lh);
#pragma unroll
            for (int nt = 0; nt < 4; ++nt) {
                v16bf bm = bfrag(&Vtile[cur][nt * 16 + lr][0], kc * 32, lh);
                o[nt] = wmma_bf16(pa, bm, o[nt]);
            }
        }
    }

    // normalize and emit OV in [B, S, H*D] layout for the output projection
#pragma unroll
    for (int nt = 0; nt < 4; ++nt) {
        int d = nt * 16 + lr;
#pragma unroll
        for (int v = 0; v < 8; ++v) {
            int s = qbase + v + 8 * lh;
            float val = o[nt][v] / l[v];
            OV[(b * SLEN + s) * (NH * HD) + h * HD + d] = (__bf16)val;
        }
    }
}

// ---------------- kernel 4: output projection ----------------
// OV [B*S, 1024] bf16  x  WOt [E][1024] bf16 (+b_O) -> out [B*S, E] fp32
__global__ __launch_bounds__(128) void outproj_kernel(const __bf16* __restrict__ OVm,
                                                      const __bf16* __restrict__ WOt,
                                                      const float* __restrict__ bO,
                                                      float* __restrict__ out) {
    int tid = threadIdx.x;
    int w = tid >> 5, lane = tid & 31, lr = lane & 15, lh = lane >> 4;
    int rbase = blockIdx.y * 64 + w * 16;  // rows over B*S
    int ebase = blockIdx.x * 64;

    const __bf16* arow = OVm + (rbase + lr) * (NH * HD);

    v8f acc[4];
#pragma unroll
    for (int nt = 0; nt < 4; ++nt) acc[nt] = zero8();

    for (int k0 = 0; k0 < NH * HD; k0 += 32) {
        v16bf a = afrag(arow, k0, lh);
#pragma unroll
        for (int nt = 0; nt < 4; ++nt) {
            v16bf bm = bfrag(WOt + (ebase + nt * 16 + lr) * (NH * HD), k0, lh);
            acc[nt] = wmma_bf16(a, bm, acc[nt]);
        }
    }

#pragma unroll
    for (int nt = 0; nt < 4; ++nt) {
        int e = ebase + nt * 16 + lr;
        float bv = bO[e];
#pragma unroll
        for (int v = 0; v < 8; ++v) {
            int row = rbase + v + 8 * lh;
            out[row * EMB + e] = acc[nt][v] + bv;
        }
    }
}

// ---------------- launch ----------------
extern "C" void kernel_launch(void* const* d_in, const int* in_sizes, int n_in,
                              void* d_out, int out_size, void* d_ws, size_t ws_size,
                              hipStream_t stream) {
    const float* query = (const float*)d_in[0];
    const float* key   = (const float*)d_in[1];
    const float* value = (const float*)d_in[2];
    const float* WQ = (const float*)d_in[3];
    const float* WK = (const float*)d_in[4];
    const float* WV = (const float*)d_in[5];
    const float* WO = (const float*)d_in[6];
    const float* bQ = (const float*)d_in[7];
    const float* bK = (const float*)d_in[8];
    const float* bV = (const float*)d_in[9];
    const float* bO = (const float*)d_in[10];
    float* out = (float*)d_out;

    char* ws = (char*)d_ws;
    size_t off = 0;
    const size_t wbytes  = (size_t)NH * HD * EMB * sizeof(__bf16);           // 2 MB
    const size_t qkbytes = (size_t)BATCH * NH * SLEN * HD * sizeof(__bf16);  // 16.8 MB

    __bf16* wtq = (__bf16*)(ws + off); off += wbytes;
    __bf16* wtk = (__bf16*)(ws + off); off += wbytes;
    __bf16* wtv = (__bf16*)(ws + off); off += wbytes;
    __bf16* wot = (__bf16*)(ws + off); off += wbytes;
    __bf16* Qp  = (__bf16*)(ws + off); off += qkbytes;
    __bf16* Kp  = (__bf16*)(ws + off); off += qkbytes;
    __bf16* Vtp = (__bf16*)(ws + off); off += qkbytes;
    __bf16* OVp = (__bf16*)(ws + off); off += qkbytes;

    // 1) weight transposes to bf16
    {
        int n = NH * EMB * HD;
        cvt_weights<<<(n + 255) / 256, 256, 0, stream>>>(WQ, WK, WV, WO, wtq, wtk, wtv, wot);
    }
    // 2) Q/K/V projections (V stored transposed per head)
    dim3 gp(SLEN / 64, BATCH * NH);
    proj_kernel<<<gp, 128, 0, stream>>>(query, wtq, bQ, Qp, 0);
    proj_kernel<<<gp, 128, 0, stream>>>(key,   wtk, bK, Kp, 0);
    proj_kernel<<<gp, 128, 0, stream>>>(value, wtv, bV, Vtp, 1);
    // 3) flash attention (async double-buffered K/V tiles)
    attn_kernel<<<gp, 128, 0, stream>>>(Qp, Kp, Vtp, OVp);
    // 4) output projection + bias
    dim3 go(EMB / 64, (BATCH * SLEN) / 64);
    outproj_kernel<<<go, 128, 0, stream>>>(OVp, wot, bO, out);
}
